// ConformerDecoder_81389630259589
// MI455X (gfx1250) — matrix operand order repaired and checked by reading, not compiled, not run on hardware
//
#include <hip/hip_runtime.h>
#include <hip/hip_bf16.h>
#include <math.h>

#define HID    640
#define H4     2560
#define NHEAD  8
#define DHEAD  80
#define BATCH  32
#define TENC   512
#define VOCAB  8192
#define MAXLEN 128

typedef __attribute__((ext_vector_type(16))) __bf16 v16bf;
typedef __attribute__((ext_vector_type(8)))  __bf16 v8bf;
typedef __attribute__((ext_vector_type(8)))  float  v8f;

__device__ __forceinline__ float sigmf_(float x) { return 1.f / (1.f + __expf(-x)); }

// ---------------- fp32 -> bf16 conversion (grid-stride) ----------------
__global__ void cvt_bf16_kernel(const float* __restrict__ s, __bf16* __restrict__ d, long n) {
  long stride = (long)gridDim.x * blockDim.x;
  for (long i = blockIdx.x * (long)blockDim.x + threadIdx.x; i < n; i += stride)
    d[i] = (__bf16)s[i];
}

// ---------------- state init: h=c=0, tok=SOS ----------------
__global__ void init_state_kernel(float* c0, float* c1, __bf16* h0, __bf16* cat, int* tok) {
  int i = blockIdx.x * blockDim.x + threadIdx.x;
  if (i < BATCH * HID) { c0[i] = 0.f; c1[i] = 0.f; h0[i] = (__bf16)0.f; }
  if (i < BATCH * 2 * HID) cat[i] = (__bf16)0.f;
  if (i < BATCH) tok[i] = 1;  // SOS
}

// ---------------- embedding gather ----------------
__global__ void embed_gather_kernel(const __bf16* __restrict__ emb, const int* __restrict__ tok,
                                    __bf16* __restrict__ x) {
  int i = blockIdx.x * blockDim.x + threadIdx.x;
  if (i >= BATCH * HID) return;
  int b = i / HID, j = i - b * HID;
  x[i] = emb[(size_t)tok[b] * HID + j];
}

// ---------------- WMMA GEMM: C[M,N] = A[M,K](bf16) @ W[N,K]^T(bf16) (+A2@W2^T) + bias ----------------
// One wave computes a 16x64 slab: ONE A fragment is amortized over FOUR B fragments / WMMAs per
// 32-wide K step (load:wmma ratio 1.5 instead of 3 -> loop becomes WMMA-issue, not VMEM, limited).
// A-fragment: lane l -> row l%16, 16B chunks at K+16*(l/16) and K+16*(l/16)+8.
// B-fragment: lane l -> W row l%16 (== B column), 32B contiguous at K+16*(l/16).
__device__ __forceinline__ v16bf load_afrag(const __bf16* base, int kb) {
  union { v16bf v; v8bf h[2]; } u;
  u.h[0] = *(const v8bf*)(base + kb);
  u.h[1] = *(const v8bf*)(base + kb + 8);
  return u.v;
}

__global__ __launch_bounds__(128) void wmma_gemm_bt_kernel(
    const __bf16* __restrict__ A, int lda,
    const __bf16* __restrict__ W, int ldb, int K,
    const __bf16* __restrict__ A2, int lda2,
    const __bf16* __restrict__ W2, int ldb2, int K2,
    const float* __restrict__ bias1, const float* __restrict__ bias2,
    float* __restrict__ C, int ldc,
    __bf16* __restrict__ Cbf, int ldcbf,
    int N, int act) {
  const int lane = threadIdx.x & 31;
  const int wave = threadIdx.x >> 5;
  const int nt = (blockIdx.x * 4 + wave) * 64;  // 64 output columns per wave
  const int mt = blockIdx.y * 16;
  if (nt >= N) return;  // wave-uniform: EXEC stays all-1s for WMMA
  const int half = lane >> 4;
  const int lm = lane & 15;
  v8f acc0 = {0.f, 0.f, 0.f, 0.f, 0.f, 0.f, 0.f, 0.f};
  v8f acc1 = acc0, acc2 = acc0, acc3 = acc0;
  {
    const __bf16* ap = A + (size_t)(mt + lm) * lda + 16 * half;
    const __bf16* wp = W + (size_t)(nt + lm) * ldb + 16 * half;
    const size_t ws = (size_t)16 * ldb;  // stride between 16-col B fragments
    for (int kb = 0; kb < K; kb += 32) {
      __builtin_prefetch(wp + kb + 128, 0, 1);  // -> global_prefetch_b8
      v16bf av = load_afrag(ap, kb);
      v16bf b0 = *(const v16bf*)(wp + kb);
      v16bf b1 = *(const v16bf*)(wp + ws + kb);
      v16bf b2 = *(const v16bf*)(wp + 2 * ws + kb);
      v16bf b3 = *(const v16bf*)(wp + 3 * ws + kb);
      acc0 = __builtin_amdgcn_wmma_f32_16x16x32_bf16(false, av, false, b0, (short)0, acc0, false, false);
      acc1 = __builtin_amdgcn_wmma_f32_16x16x32_bf16(false, av, false, b1, (short)0, acc1, false, false);
      acc2 = __builtin_amdgcn_wmma_f32_16x16x32_bf16(false, av, false, b2, (short)0, acc2, false, false);
      acc3 = __builtin_amdgcn_wmma_f32_16x16x32_bf16(false, av, false, b3, (short)0, acc3, false, false);
    }
  }
  if (A2) {  // fused second GEMM (LSTM: x@Wih^T + h@Whh^T)
    const __bf16* ap = A2 + (size_t)(mt + lm) * lda2 + 16 * half;
    const __bf16* wp = W2 + (size_t)(nt + lm) * ldb2 + 16 * half;
    const size_t ws = (size_t)16 * ldb2;
    for (int kb = 0; kb < K2; kb += 32) {
      __builtin_prefetch(wp + kb + 128, 0, 1);
      v16bf av = load_afrag(ap, kb);
      v16bf b0 = *(const v16bf*)(wp + kb);
      v16bf b1 = *(const v16bf*)(wp + ws + kb);
      v16bf b2 = *(const v16bf*)(wp + 2 * ws + kb);
      v16bf b3 = *(const v16bf*)(wp + 3 * ws + kb);
      acc0 = __builtin_amdgcn_wmma_f32_16x16x32_bf16(false, av, false, b0, (short)0, acc0, false, false);
      acc1 = __builtin_amdgcn_wmma_f32_16x16x32_bf16(false, av, false, b1, (short)0, acc1, false, false);
      acc2 = __builtin_amdgcn_wmma_f32_16x16x32_bf16(false, av, false, b2, (short)0, acc2, false, false);
      acc3 = __builtin_amdgcn_wmma_f32_16x16x32_bf16(false, av, false, b3, (short)0, acc3, false, false);
    }
  }
  // epilogue: 4 16x16 tiles
  v8f accs[4] = {acc0, acc1, acc2, acc3};
#pragma unroll
  for (int t = 0; t < 4; ++t) {
    const int col = nt + t * 16 + lm;
    float bvv = 0.f;
    if (bias1) bvv += bias1[col];
    if (bias2) bvv += bias2[col];
#pragma unroll
    for (int j = 0; j < 8; ++j) {
      int row = mt + j + 8 * half;
      float v = accs[t][j] + bvv;
      if (act == 1) v = tanhf(v);
      C[(size_t)row * ldc + col] = v;
      if (Cbf) Cbf[(size_t)row * ldcbf + col] = (__bf16)v;
    }
  }
}

// ---------------- LSTM elementwise: gates [B, 4H] (i,f,g,o) + c -> new c (f32), h (bf16) ----------------
__global__ void lstm_ew_kernel(const float* __restrict__ G, float* __restrict__ c,
                               __bf16* __restrict__ h, int ldh) {
  int i = blockIdx.x * blockDim.x + threadIdx.x;
  if (i >= BATCH * HID) return;
  int b = i / HID, j = i - b * HID;
  const float* g = G + (size_t)b * H4;
  float ig = sigmf_(g[j]);
  float fg = sigmf_(g[HID + j]);
  float gg = tanhf(g[2 * HID + j]);
  float og = sigmf_(g[3 * HID + j]);
  float cv = fg * c[i] + ig * gg;
  c[i] = cv;
  h[(size_t)b * ldh + j] = (__bf16)(og * tanhf(cv));
}

// ---------------- attention: one block per (b, head); ctx written as bf16 into cat[:, 640:] ----------------
__global__ __launch_bounds__(128) void attn_kernel(const float* __restrict__ q,
                                                   const float* __restrict__ Kf,
                                                   const float* __restrict__ Vf,
                                                   __bf16* __restrict__ cat) {
  __shared__ float sc[TENC];
  __shared__ float red[128];
  __shared__ float qs[DHEAD];
  int bh = blockIdx.x;
  int b = bh / NHEAD, hh = bh - b * NHEAD;
  int tid = threadIdx.x;
  if (tid < DHEAD) qs[tid] = q[b * HID + hh * DHEAD + tid];
  __syncthreads();
  float lmax = -3.0e38f;
  for (int t = tid; t < TENC; t += 128) {
    const float* kp = Kf + ((size_t)b * TENC + t) * HID + hh * DHEAD;
    float s = 0.f;
#pragma unroll 8
    for (int d = 0; d < DHEAD; ++d) s += qs[d] * kp[d];
    s *= 0.11180339887498949f;  // 1/sqrt(80)
    sc[t] = s;
    lmax = fmaxf(lmax, s);
  }
  red[tid] = lmax; __syncthreads();
  for (int s = 64; s > 0; s >>= 1) { if (tid < s) red[tid] = fmaxf(red[tid], red[tid + s]); __syncthreads(); }
  float m = red[0]; __syncthreads();
  float lsum = 0.f;
  for (int t = tid; t < TENC; t += 128) { float e = __expf(sc[t] - m); sc[t] = e; lsum += e; }
  red[tid] = lsum; __syncthreads();
  for (int s = 64; s > 0; s >>= 1) { if (tid < s) red[tid] += red[tid + s]; __syncthreads(); }
  float inv = 1.f / red[0];
  for (int d = tid; d < DHEAD; d += 128) {
    float acc = 0.f;
    const float* vp = Vf + (size_t)b * TENC * HID + hh * DHEAD + d;
    for (int t = 0; t < TENC; ++t) acc += sc[t] * vp[(size_t)t * HID];
    cat[(size_t)b * (2 * HID) + HID + hh * DHEAD + d] = (__bf16)(acc * inv);
  }
}

// ---------------- log_softmax + greedy argmax; one block per batch row ----------------
__global__ __launch_bounds__(256) void lsm_argmax_kernel(const float* __restrict__ logits,
                                                         float* __restrict__ out,
                                                         int* __restrict__ tok, int step) {
  __shared__ float rv[256];
  __shared__ int ri[256];
  int b = blockIdx.x, tid = threadIdx.x;
  const float* lg = logits + (size_t)b * VOCAB;
  float m = -3.0e38f; int mi = 0;
  for (int v = tid; v < VOCAB; v += 256) { float x = lg[v]; if (x > m) { m = x; mi = v; } }
  rv[tid] = m; ri[tid] = mi; __syncthreads();
  for (int s = 128; s > 0; s >>= 1) {
    if (tid < s) {
      if (rv[tid + s] > rv[tid] || (rv[tid + s] == rv[tid] && ri[tid + s] < ri[tid])) {
        rv[tid] = rv[tid + s]; ri[tid] = ri[tid + s];
      }
    }
    __syncthreads();
  }
  float M = rv[0]; int arg = ri[0]; __syncthreads();
  float sum = 0.f;
  for (int v = tid; v < VOCAB; v += 256) sum += __expf(lg[v] - M);
  rv[tid] = sum; __syncthreads();
  for (int s = 128; s > 0; s >>= 1) { if (tid < s) rv[tid] += rv[tid + s]; __syncthreads(); }
  float lse = M + __logf(rv[0]);
  float* op = out + ((size_t)b * MAXLEN + step) * VOCAB;
  for (int v = tid; v < VOCAB; v += 256) op[v] = lg[v] - lse;
  if (tid == 0) tok[b] = arg;
}

// ==================== host launcher ====================
extern "C" void kernel_launch(void* const* d_in, const int* in_sizes, int n_in,
                              void* d_out, int out_size, void* d_ws, size_t ws_size,
                              hipStream_t stream) {
  (void)in_sizes; (void)n_in; (void)out_size; (void)ws_size;
  const float* enc  = (const float*)d_in[0];
  const float* emb  = (const float*)d_in[1];
  const float* Wq   = (const float*)d_in[2];  const float* bq   = (const float*)d_in[3];
  const float* Wk   = (const float*)d_in[4];  const float* bk   = (const float*)d_in[5];
  const float* Wv   = (const float*)d_in[6];  const float* bv   = (const float*)d_in[7];
  const float* Wih0 = (const float*)d_in[8];  const float* Whh0 = (const float*)d_in[9];
  const float* bih0 = (const float*)d_in[10]; const float* bhh0 = (const float*)d_in[11];
  const float* Wih1 = (const float*)d_in[12]; const float* Whh1 = (const float*)d_in[13];
  const float* bih1 = (const float*)d_in[14]; const float* bhh1 = (const float*)d_in[15];
  const float* fc1w = (const float*)d_in[16]; const float* fc1b = (const float*)d_in[17];
  const float* fc2w = (const float*)d_in[18]; const float* fc2b = (const float*)d_in[19];
  float* out = (float*)d_out;

  // ---- workspace carve (~146 MB) ----
  char* p = (char*)d_ws;
  auto carve = [&](size_t bytes) -> void* {
    void* r = (void*)p; p += (bytes + 255) & ~(size_t)255; return r;
  };
  __bf16* encB  = (__bf16*)carve((size_t)BATCH * TENC * HID * 2);
  __bf16* embB  = (__bf16*)carve((size_t)VOCAB * HID * 2);
  __bf16* WqB   = (__bf16*)carve((size_t)HID * HID * 2);
  __bf16* WkB   = (__bf16*)carve((size_t)HID * HID * 2);
  __bf16* WvB   = (__bf16*)carve((size_t)HID * HID * 2);
  __bf16* Wih0B = (__bf16*)carve((size_t)H4 * HID * 2);
  __bf16* Whh0B = (__bf16*)carve((size_t)H4 * HID * 2);
  __bf16* Wih1B = (__bf16*)carve((size_t)H4 * HID * 2);
  __bf16* Whh1B = (__bf16*)carve((size_t)H4 * HID * 2);
  __bf16* fc1B  = (__bf16*)carve((size_t)HID * 2 * HID * 2);
  __bf16* fc2B  = (__bf16*)carve((size_t)VOCAB * HID * 2);
  float*  Kf    = (float*)carve((size_t)BATCH * TENC * HID * 4);
  float*  Vf    = (float*)carve((size_t)BATCH * TENC * HID * 4);
  float*  G     = (float*)carve((size_t)BATCH * H4 * 4);
  float*  c0    = (float*)carve((size_t)BATCH * HID * 4);
  float*  c1    = (float*)carve((size_t)BATCH * HID * 4);
  __bf16* h0B   = (__bf16*)carve((size_t)BATCH * HID * 2);
  __bf16* xB    = (__bf16*)carve((size_t)BATCH * HID * 2);
  __bf16* catB  = (__bf16*)carve((size_t)BATCH * 2 * HID * 2);  // [h1 | ctx] bf16
  float*  qf    = (float*)carve((size_t)BATCH * HID * 4);
  float*  zf    = (float*)carve((size_t)BATCH * HID * 4);
  __bf16* zB    = (__bf16*)carve((size_t)BATCH * HID * 2);
  float*  logit = (float*)carve((size_t)BATCH * VOCAB * 4);
  int*    tok   = (int*)carve((size_t)BATCH * 4);

  auto cvt = [&](const float* s, __bf16* d, size_t n) {
    int blocks = (int)((n + 255) / 256); if (blocks > 16384) blocks = 16384;
    cvt_bf16_kernel<<<dim3(blocks), dim3(256), 0, stream>>>(s, d, (long)n);
  };
  cvt(enc,  encB,  (size_t)BATCH * TENC * HID);
  cvt(emb,  embB,  (size_t)VOCAB * HID);
  cvt(Wq,   WqB,   (size_t)HID * HID);
  cvt(Wk,   WkB,   (size_t)HID * HID);
  cvt(Wv,   WvB,   (size_t)HID * HID);
  cvt(Wih0, Wih0B, (size_t)H4 * HID);
  cvt(Whh0, Whh0B, (size_t)H4 * HID);
  cvt(Wih1, Wih1B, (size_t)H4 * HID);
  cvt(Whh1, Whh1B, (size_t)H4 * HID);
  cvt(fc1w, fc1B,  (size_t)HID * 2 * HID);
  cvt(fc2w, fc2B,  (size_t)VOCAB * HID);

  auto gemm = [&](const __bf16* A, int lda, const __bf16* W, int ldb, int K,
                  const __bf16* A2, int lda2, const __bf16* W2, int ldb2, int K2,
                  const float* b1, const float* b2, float* C, int ldc,
                  __bf16* Cbf, int ldcbf, int M, int N, int act) {
    dim3 grid((N / 64 + 3) / 4, M / 16), blk(128);  // 64 cols per wave, 4 waves per block
    wmma_gemm_bt_kernel<<<grid, blk, 0, stream>>>(A, lda, W, ldb, K, A2, lda2, W2, ldb2, K2,
                                                  b1, b2, C, ldc, Cbf, ldcbf, N, act);
  };

  // ---- loop-invariant K/V projections: [B*T,HID] = encB @ Wk^T / Wv^T ----
  gemm(encB, HID, WkB, HID, HID, nullptr, 0, nullptr, 0, 0, bk, nullptr,
       Kf, HID, nullptr, 0, BATCH * TENC, HID, 0);
  gemm(encB, HID, WvB, HID, HID, nullptr, 0, nullptr, 0, 0, bv, nullptr,
       Vf, HID, nullptr, 0, BATCH * TENC, HID, 0);

  init_state_kernel<<<dim3(160), dim3(256), 0, stream>>>(c0, c1, h0B, catB, tok);

  // ---- 128 sequential greedy decode steps ----
  for (int step = 0; step < MAXLEN; ++step) {
    embed_gather_kernel<<<dim3(80), dim3(256), 0, stream>>>(embB, tok, xB);
    // LSTM layer 0 gates: x@Wih0^T + h0@Whh0^T (fused)
    gemm(xB, HID, Wih0B, HID, HID, h0B, HID, Whh0B, HID, HID, bih0, bhh0,
         G, H4, nullptr, 0, BATCH, H4, 0);
    lstm_ew_kernel<<<dim3(80), dim3(256), 0, stream>>>(G, c0, h0B, HID);
    // LSTM layer 1 gates: h0@Wih1^T + h1@Whh1^T (h1 lives in cat[:, :640], stride 1280)
    gemm(h0B, HID, Wih1B, HID, HID, catB, 2 * HID, Whh1B, HID, HID, bih1, bhh1,
         G, H4, nullptr, 0, BATCH, H4, 0);
    lstm_ew_kernel<<<dim3(80), dim3(256), 0, stream>>>(G, c1, catB, 2 * HID);
    // q = h1 @ Wq^T + bq
    gemm(catB, 2 * HID, WqB, HID, HID, nullptr, 0, nullptr, 0, 0, bq, nullptr,
         qf, HID, nullptr, 0, BATCH, HID, 0);
    attn_kernel<<<dim3(BATCH * NHEAD), dim3(128), 0, stream>>>(qf, Kf, Vf, catB);
    // z = tanh([h1|ctx] @ fc1_w^T + fc1_b), write bf16 copy for fc2
    gemm(catB, 2 * HID, fc1B, 2 * HID, 2 * HID, nullptr, 0, nullptr, 0, 0, fc1b, nullptr,
         zf, HID, zB, HID, BATCH, HID, 1);
    // logits = z @ fc2_w^T + fc2_b
    gemm(zB, HID, fc2B, HID, HID, nullptr, 0, nullptr, 0, 0, fc2b, nullptr,
         logit, VOCAB, nullptr, 0, BATCH, VOCAB, 0);
    lsm_argmax_kernel<<<dim3(BATCH), dim3(256), 0, stream>>>(logit, out, tok, step);
  }
}
